// ExpertFFN_48627619726050
// MI455X (gfx1250) — compile-verified, matrix operand order checked by reading, not run on hardware
//
#include <hip/hip_runtime.h>
#include <hip/hip_bf16.h>
#include <math.h>

// ---------------------------------------------------------------------------
// Top-1 MoE FFN for MI455X (gfx1250, wave32, WMMA bf16 16x16x32).
// Only the selected expert runs per token (8x less work than the reference's
// dense-all-experts formulation; outputs identical).
// Grouped-GEMM tiling: 32 tokens x 512 cols per block, 8 waves, each wave
// 2 m-tiles x 4 n-tiles (8 WMMAs / K-step), double-buffered LDS staging,
// loop-invariant addressing hoisted out of the K-loop.
// ---------------------------------------------------------------------------

typedef __attribute__((ext_vector_type(16))) __bf16 v16bf;
typedef __attribute__((ext_vector_type(8)))  float  v8f;

#define NE     8           // experts
#define DD     1024        // hidden
#define FF     4096        // ffn dim
#define NTOK   4096        // B*T = 2*2048
#define MT     32          // tokens per tile
#define MAXT32 (NTOK/MT)   // max 32-token tiles per expert

// ---------------------------- kernel 0: init --------------------------------
__global__ void moe_zero_counts(int* __restrict__ counts) {
    if (threadIdx.x < NE) counts[threadIdx.x] = 0;
}

// ---------------------------- kernel 1: router -------------------------------
// one block per token; wave w computes expert w's logit (dot over D=1024)
__global__ __launch_bounds__(256)
void moe_router_kernel(const float* __restrict__ x,
                       const float* __restrict__ rW,
                       const float* __restrict__ rb,
                       float* __restrict__ probs_out,
                       float* __restrict__ logits_out,
                       int* __restrict__ counts,
                       int* __restrict__ bucket) {
    const int tok  = blockIdx.x;
    const int tid  = threadIdx.x;
    __shared__ float xs[DD];
    __shared__ float lg[NE];
    for (int i = tid; i < DD; i += 256) xs[i] = x[(size_t)tok * DD + i];
    __syncthreads();

    const int w    = tid >> 5;   // wave id == expert id (wave32)
    const int lane = tid & 31;
    const float* wrow = rW + (size_t)w * DD;
    float p = 0.f;
    for (int d = lane; d < DD; d += 32) p += xs[d] * wrow[d];
    #pragma unroll
    for (int off = 16; off > 0; off >>= 1) p += __shfl_down(p, off, 32);
    if (lane == 0) lg[w] = p + rb[w];
    __syncthreads();

    if (tid == 0) {
        float mx = lg[0]; int am = 0;
        #pragma unroll
        for (int e = 1; e < NE; ++e) if (lg[e] > mx) { mx = lg[e]; am = e; }
        float pr[NE]; float s = 0.f;
        #pragma unroll
        for (int e = 0; e < NE; ++e) { pr[e] = __expf(lg[e] - mx); s += pr[e]; }
        const float inv = 1.f / s;
        #pragma unroll
        for (int e = 0; e < NE; ++e) {
            probs_out [(size_t)tok * NE + e] = pr[e] * inv;
            logits_out[(size_t)tok * NE + e] = lg[e];
        }
        const int slot = atomicAdd(&counts[am], 1);
        bucket[(size_t)am * NTOK + slot] = tok;
    }
}

__device__ __forceinline__ float gelu_exact(float v) {
    return 0.5f * v * (1.f + erff(v * 0.70710678118654752f));
}

// ------------------- kernel 2: H = gelu(x @ W1[e] + b1[e]) -------------------
__global__ __launch_bounds__(256)
void moe_ffn1_kernel(const float* __restrict__ x,
                     const float* __restrict__ W1,
                     const float* __restrict__ b1,
                     const int* __restrict__ counts,
                     const int* __restrict__ bucket,
                     __bf16* __restrict__ H) {
    const int e   = blockIdx.x / MAXT32;
    const int t   = blockIdx.x % MAXT32;
    const int cnt = counts[e];
    if (t * MT >= cnt) return;
    const int f0  = blockIdx.y * 512;
    const int tid = threadIdx.x;

    __shared__ alignas(32) __bf16 alds[2][MT][32];    // A: 32 tokens x 32 K
    __shared__ alignas(32) __bf16 blds[2][512][32];   // B^T: 512 cols x 32 K
    __shared__ int toks[MT];

    if (tid < MT) {
        const int slot = t * MT + tid;
        toks[tid] = (slot < cnt) ? bucket[(size_t)e * NTOK + slot] : -1;
    }
    __syncthreads();

    const int w    = tid >> 5;
    const int lane = tid & 31;
    const int m    = lane & 15;
    const int half = lane >> 4;

    // loop-invariant staging addresses (hoisted out of the K-loop)
    const int arow = tid >> 3;
    const int acol = (tid & 7) * 4;
    const int atk  = toks[arow];
    const float* __restrict__ xrow =
        x + (size_t)(atk < 0 ? 0 : atk) * DD + acol;          // + k0 per step
    const int c0 = tid * 2;
    const float* __restrict__ wcol =
        W1 + (size_t)e * DD * FF + f0 + c0;                   // + (k0+k)*FF

    v8f acc[2][4];
    {
        v8f z = {};
        #pragma unroll
        for (int i = 0; i < 2; ++i)
            #pragma unroll
            for (int j = 0; j < 4; ++j) acc[i][j] = z;
    }

    auto load_tiles = [&](int buf, int k0) {
        // A 32x32: 4 contiguous fp32 per thread -> bf16 (cvt_pk + b64 store)
        {
            const float4 v = *reinterpret_cast<const float4*>(xrow + k0);
            alds[buf][arow][acol + 0] = (__bf16)v.x;
            alds[buf][arow][acol + 1] = (__bf16)v.y;
            alds[buf][arow][acol + 2] = (__bf16)v.z;
            alds[buf][arow][acol + 3] = (__bf16)v.w;
        }
        // B: W1[e][k0+k][f0+col] -> blds[col][k] (transposed so each lane's
        // 16-K fragment is one contiguous 32B ds read); float2 along f
        #pragma unroll 8
        for (int k = 0; k < 32; ++k) {
            const float2 v = *reinterpret_cast<const float2*>(
                wcol + (size_t)(k0 + k) * FF);
            blds[buf][c0    ][k] = (__bf16)v.x;
            blds[buf][c0 + 1][k] = (__bf16)v.y;
        }
    };

    load_tiles(0, 0);
    __syncthreads();

    for (int s = 0; s < DD / 32; ++s) {
        const int buf = s & 1;
        if (s + 1 < DD / 32) load_tiles(buf ^ 1, (s + 1) * 32);

        // contiguous k order (A and B alike): lanes 0-15 hold K=0..15, lanes 16-31 hold K=16..31
        v16bf a[2], b[4];
        #pragma unroll
        for (int i = 0; i < 2; ++i)
            a[i] = *reinterpret_cast<const v16bf*>(&alds[buf][i * 16 + m][half * 16]);
        #pragma unroll
        for (int j = 0; j < 4; ++j)
            b[j] = *reinterpret_cast<const v16bf*>(&blds[buf][w * 64 + j * 16 + m][half * 16]);
        #pragma unroll
        for (int i = 0; i < 2; ++i)
            #pragma unroll
            for (int j = 0; j < 4; ++j)
                acc[i][j] = __builtin_amdgcn_wmma_f32_16x16x32_bf16(
                    false, a[i], false, b[j], (short)0, acc[i][j], false, false);
        __syncthreads();
    }

    // epilogue: bias + exact GELU, store bf16 H rows (D-matrix layout:
    // VGPR r, lanes 0-15 -> M=r, lanes 16-31 -> M=r+8, N=lane&15)
    const float* b1e = b1 + (size_t)e * FF;
    #pragma unroll
    for (int i = 0; i < 2; ++i) {
        const int rb0 = i * 16 + half * 8;
        #pragma unroll
        for (int r = 0; r < 8; ++r) {
            const int tk = toks[rb0 + r];
            if (tk < 0) continue;
            #pragma unroll
            for (int j = 0; j < 4; ++j) {
                const int col = f0 + w * 64 + j * 16 + m;
                const float v = acc[i][j][r] + b1e[col];
                H[(size_t)tk * FF + col] = (__bf16)gelu_exact(v);
            }
        }
    }
}

// ------------------- kernel 3: out = H @ W2[e] + b2[e] -----------------------
__global__ __launch_bounds__(256)
void moe_ffn2_kernel(const __bf16* __restrict__ H,
                     const float* __restrict__ W2,
                     const float* __restrict__ b2,
                     const int* __restrict__ counts,
                     const int* __restrict__ bucket,
                     float* __restrict__ out) {
    const int e   = blockIdx.x / MAXT32;
    const int t   = blockIdx.x % MAXT32;
    const int cnt = counts[e];
    if (t * MT >= cnt) return;
    const int d0  = blockIdx.y * 512;
    const int tid = threadIdx.x;

    __shared__ alignas(32) __bf16 alds[2][MT][32];
    __shared__ alignas(32) __bf16 blds[2][512][32];
    __shared__ int toks[MT];

    if (tid < MT) {
        const int slot = t * MT + tid;
        toks[tid] = (slot < cnt) ? bucket[(size_t)e * NTOK + slot] : -1;
    }
    __syncthreads();

    const int w    = tid >> 5;
    const int lane = tid & 31;
    const int m    = lane & 15;
    const int half = lane >> 4;

    const int arow = tid >> 3;
    const int acol = (tid & 7) * 4;
    const int atk  = toks[arow];
    const __bf16* __restrict__ hrow =
        H + (size_t)(atk < 0 ? 0 : atk) * FF + acol;          // + k0 per step
    const int c0 = tid * 2;
    const float* __restrict__ wcol =
        W2 + (size_t)e * FF * DD + d0 + c0;                   // + (k0+k)*DD

    v8f acc[2][4];
    {
        v8f z = {};
        #pragma unroll
        for (int i = 0; i < 2; ++i)
            #pragma unroll
            for (int j = 0; j < 4; ++j) acc[i][j] = z;
    }

    auto load_tiles = [&](int buf, int k0) {
        // A 32x32 from H (already bf16): 8-byte raw copy
        {
            const ushort4 v = *reinterpret_cast<const ushort4*>(hrow + k0);
            *reinterpret_cast<ushort4*>(&alds[buf][arow][acol]) = v;
        }
        // B: W2[e][k0+k][d0+col] -> blds[col][k]
        #pragma unroll 8
        for (int k = 0; k < 32; ++k) {
            const float2 v = *reinterpret_cast<const float2*>(
                wcol + (size_t)(k0 + k) * DD);
            blds[buf][c0    ][k] = (__bf16)v.x;
            blds[buf][c0 + 1][k] = (__bf16)v.y;
        }
    };

    load_tiles(0, 0);
    __syncthreads();

    for (int s = 0; s < FF / 32; ++s) {
        const int buf = s & 1;
        if (s + 1 < FF / 32) load_tiles(buf ^ 1, (s + 1) * 32);

        v16bf a[2], b[4];
        #pragma unroll
        for (int i = 0; i < 2; ++i)
            a[i] = *reinterpret_cast<const v16bf*>(&alds[buf][i * 16 + m][half * 16]);
        #pragma unroll
        for (int j = 0; j < 4; ++j)
            b[j] = *reinterpret_cast<const v16bf*>(&blds[buf][w * 64 + j * 16 + m][half * 16]);
        #pragma unroll
        for (int i = 0; i < 2; ++i)
            #pragma unroll
            for (int j = 0; j < 4; ++j)
                acc[i][j] = __builtin_amdgcn_wmma_f32_16x16x32_bf16(
                    false, a[i], false, b[j], (short)0, acc[i][j], false, false);
        __syncthreads();
    }

    const float* b2e = b2 + (size_t)e * DD;
    #pragma unroll
    for (int i = 0; i < 2; ++i) {
        const int rb0 = i * 16 + half * 8;
        #pragma unroll
        for (int r = 0; r < 8; ++r) {
            const int tk = toks[rb0 + r];
            if (tk < 0) continue;
            #pragma unroll
            for (int j = 0; j < 4; ++j) {
                const int col = d0 + w * 64 + j * 16 + m;
                out[(size_t)tk * DD + col] = acc[i][j][r] + b2e[col];
            }
        }
    }
}

// ----------------------------- launcher --------------------------------------
extern "C" void kernel_launch(void* const* d_in, const int* in_sizes, int n_in,
                              void* d_out, int out_size, void* d_ws, size_t ws_size,
                              hipStream_t stream) {
    const float* x   = (const float*)d_in[0];   // [2,2048,1024]
    const float* rW  = (const float*)d_in[1];   // [8,1024]
    const float* rb  = (const float*)d_in[2];   // [8]
    const float* W1  = (const float*)d_in[3];   // [8,1024,4096]
    const float* b1  = (const float*)d_in[4];   // [8,4096]
    const float* W2  = (const float*)d_in[5];   // [8,4096,1024]
    const float* b2  = (const float*)d_in[6];   // [8,1024]

    float* out        = (float*)d_out;
    float* expert_out = out;                                    // [N, D]
    float* probs      = out + (size_t)NTOK * DD;                // [N, E]
    float* logits     = probs + (size_t)NTOK * NE;              // [N, E]

    char*   ws     = (char*)d_ws;
    int*    counts = (int*)ws;                                  // E ints
    int*    bucket = (int*)(ws + 128);                          // E*N ints
    size_t  hoff   = ((size_t)128 + (size_t)NE * NTOK * 4 + 255) & ~(size_t)255;
    __bf16* H      = (__bf16*)(ws + hoff);                      // [N, F] bf16

    moe_zero_counts<<<1, 32, 0, stream>>>(counts);
    moe_router_kernel<<<NTOK, 256, 0, stream>>>(x, rW, rb, probs, logits, counts, bucket);
    moe_ffn1_kernel<<<dim3(NE * MAXT32, FF / 512), 256, 0, stream>>>(x, W1, b1, counts, bucket, H);
    moe_ffn2_kernel<<<dim3(NE * MAXT32, DD / 512), 256, 0, stream>>>(H, W2, b2, counts, bucket, expert_out);
}
